// VladPooling_77421080478356
// MI455X (gfx1250) — matrix operand order repaired and checked by reading, not compiled, not run on hardware
//
#include <hip/hip_runtime.h>

// ---------------------------------------------------------------------------
// VLAD pooling for MI455X (gfx1250, wave32, WMMA).
//   B=16, N=T*H*W=6272, C=512, K=64
// Pass 1: logits = x @ Wk + bias  (bf16 WMMA, f32 acc), softmax over K,
//         store assgn^T as bf16 in workspace, layout [b][k][n].
// Pass 2: weighted = assgn^T @ x  (bf16 WMMA, f32 acc); assgn row-sums are
//         computed on the matrix pipe as assgn^T @ ones (2nd WMMA per step);
//         fused epilogue (weighted - rowsum*centers)/N -> out fp32.
//
// Fragment element mapping (ISA 7.12.2, wave32, j = 2v+e):
//   A (16x32, 16-bit): K = 8h + j (j<8),  K = 16 + 8h + (j-8) (j>=8)
//       -> two contiguous 8-element runs per lane.
//   B (32x16, 16-bit): same K order as A -> two 8-element runs per lane
//       when read from transposed Wk (pass 2 uses K = 16h + j for both A and B).
//   C/D (16x16, f32):  row M = v + 8h, col N = lane&15.
// ---------------------------------------------------------------------------

typedef __attribute__((ext_vector_type(16))) __bf16 v16bf;
typedef __attribute__((ext_vector_type(8)))  __bf16 v8bf;
typedef __attribute__((ext_vector_type(8)))  float  v8f;
typedef __attribute__((ext_vector_type(4)))  float  v4f;

#define VB   16
#define VN   6272      // 32*14*14
#define VC   512
#define VK   64
#define NT16 (VN / 16) // 392
#define WKP  520       // padded LDS row stride (elements); 1040 B, 16B-aligned

#define SHUF16(lo, hi) __builtin_shufflevector((lo), (hi), 0, 1, 2, 3, 4, 5, 6, 7, \
                                               8, 9, 10, 11, 12, 13, 14, 15)

// -------------------------------------------------------------------------
// Pass 1: one wave handles one (b, 16-row n-tile).
// Wk is staged in LDS transposed+bf16: wk_t[k][c] so B fragments are
// two 16-byte runs -> ds_load_b128 pairs.
// -------------------------------------------------------------------------
__global__ __launch_bounds__(256) void vlad_pass1(
    const float* __restrict__ x,       // (B,N,C)
    const float* __restrict__ wk,      // (C,K)
    const float* __restrict__ bias,    // (K)
    __bf16* __restrict__ assgn_t)      // (B,K,N) bf16
{
    __shared__ __bf16 wk_t[VK * WKP];  // 65 KB, transposed Wk

    // cooperative fp32 -> bf16 transpose staging (reads coalesced)
    for (int i = threadIdx.x; i < VC * VK; i += blockDim.x) {
        const int c = i >> 6;          // 0..511
        const int k = i & 63;          // 0..63
        wk_t[k * WKP + c] = (__bf16)wk[i];
    }
    __syncthreads();

    const int gwave = (blockIdx.x * blockDim.x + threadIdx.x) >> 5;
    const int lane  = threadIdx.x & 31;
    const int b     = gwave / NT16;
    const int tile  = gwave % NT16;
    if (b >= VB) return;

    const int row = lane & 15;   // A-row / D-column lane index
    const int h   = lane >> 4;   // lane half

    const float* xb = x + ((size_t)b * VN + (size_t)tile * 16) * VC;

    v8f acc[4] = {};             // 16x64 logits tile (4 x 16x16)

    for (int c0 = 0; c0 < VC; c0 += 32) {
        // ---- A fragment: x row (M = n_local), two runs of 8 consecutive c
        const float* xr = xb + (size_t)row * VC + c0 + 8 * h;
        const v4f f0 = *(const v4f*)(xr);
        const v4f f1 = *(const v4f*)(xr + 4);
        const v4f f2 = *(const v4f*)(xr + 16);
        const v4f f3 = *(const v4f*)(xr + 20);
        v16bf a;
#pragma unroll
        for (int j = 0; j < 4; ++j) {
            a[j]      = (__bf16)f0[j];
            a[4 + j]  = (__bf16)f1[j];
            a[8 + j]  = (__bf16)f2[j];
            a[12 + j] = (__bf16)f3[j];
        }
#pragma unroll
        for (int kt = 0; kt < 4; ++kt) {
            // ---- B fragment: two runs of 8 K-rows of column (kt*16+row), matching A
            const __bf16* wrow = &wk_t[(kt * 16 + row) * WKP + c0 + 8 * h];
            const v8bf blo = *(const v8bf*)(wrow);
            const v8bf bhi = *(const v8bf*)(wrow + 16);
            const v16bf bb = SHUF16(blo, bhi);
            acc[kt] = __builtin_amdgcn_wmma_f32_16x16x32_bf16(
                false, a, false, bb, (short)0, acc[kt], false, false);
        }
    }

    // ---- add bias (per cluster column = kt*16 + row)
#pragma unroll
    for (int kt = 0; kt < 4; ++kt) {
        const float bv = bias[kt * 16 + row];
#pragma unroll
        for (int v = 0; v < 8; ++v) acc[kt][v] += bv;
    }

    // ---- softmax over the 64 clusters of each row M = v + 8*h.
    // Row data lives in VGPR v across the 16 lanes of a half, over 4 kt.
#pragma unroll
    for (int v = 0; v < 8; ++v) {
        float m = acc[0][v];
#pragma unroll
        for (int kt = 1; kt < 4; ++kt) m = fmaxf(m, acc[kt][v]);
#pragma unroll
        for (int s = 1; s < 16; s <<= 1) m = fmaxf(m, __shfl_xor(m, s, 32));

        float sum = 0.f;
        float e[4];
#pragma unroll
        for (int kt = 0; kt < 4; ++kt) { e[kt] = __expf(acc[kt][v] - m); sum += e[kt]; }
#pragma unroll
        for (int s = 1; s < 16; s <<= 1) sum += __shfl_xor(sum, s, 32);

        const float inv = 1.0f / sum;
#pragma unroll
        for (int kt = 0; kt < 4; ++kt) acc[kt][v] = e[kt] * inv;
    }

    // ---- store assgn transposed: assgn_t[b][k][n], one 16-byte store per kt
    const int n_base = tile * 16 + 8 * h;
#pragma unroll
    for (int kt = 0; kt < 4; ++kt) {
        const int k = kt * 16 + row;
        v8bf st;
#pragma unroll
        for (int v = 0; v < 8; ++v) st[v] = (__bf16)acc[kt][v];
        *(v8bf*)(assgn_t + ((size_t)b * VK + k) * VN + n_base) = st;
    }
}

// -------------------------------------------------------------------------
// Pass 2: one wave per 16x16 (K,C) output tile; loop over N in WMMA steps.
// Row-sums of assgn computed on the matrix pipe: acc_rs = assgn^T @ ones,
// so every D column holds rowsum(k) -> no cross-lane reduction needed.
// -------------------------------------------------------------------------
__global__ __launch_bounds__(256) void vlad_pass2(
    const float*  __restrict__ x,        // (B,N,C)
    const __bf16* __restrict__ assgn_t,  // (B,K,N)
    const float*  __restrict__ centers,  // (K,C)
    float* __restrict__ out)             // (B,K*C)
{
    const int gwave = (blockIdx.x * blockDim.x + threadIdx.x) >> 5;
    const int lane  = threadIdx.x & 31;
    const int ct = gwave & 31;          // C tile  (32)
    const int kt = (gwave >> 5) & 3;    // K tile  (4)
    const int b  = gwave >> 7;          // batch   (16)
    if (b >= VB) return;

    const int k0 = kt * 16, c0 = ct * 16;
    const int row = lane & 15;
    const int h   = lane >> 4;

    const __bf16* ab = assgn_t + ((size_t)b * VK + k0 + row) * VN + 16 * h;
    const float*  xb = x + (size_t)b * VN * VC;

    // constant all-ones B fragment (bf16 1.0 exactly representable)
    const __bf16 one = (__bf16)1.0f;
    v16bf bones;
#pragma unroll
    for (int j = 0; j < 16; ++j) bones[j] = one;

    v8f acc    = {};   // weighted_x tile
    v8f acc_rs = {};   // rowsum tile: every column = sum_n assgn[k][n]

    for (int n0 = 0; n0 < VN; n0 += 32) {
        // ---- A fragment: assgn^T (M=cluster, Kdim=n), two 16-byte runs
        const v8bf lo = *(const v8bf*)(ab + n0);
        const v8bf hi = *(const v8bf*)(ab + n0 + 8);
        const v16bf a = SHUF16(lo, hi);

        // ---- B fragment: x column slice (Kdim=n_local, N=c), fp32 -> bf16
        const float* xc = xb + (size_t)(n0 + 16 * h) * VC + c0 + row;
        v16bf bb;
#pragma unroll
        for (int j = 0; j < 16; ++j)
            bb[j] = (__bf16)xc[(size_t)j * VC];

        acc = __builtin_amdgcn_wmma_f32_16x16x32_bf16(
            false, a, false, bb, (short)0, acc, false, false);
        acc_rs = __builtin_amdgcn_wmma_f32_16x16x32_bf16(
            false, a, false, bones, (short)0, acc_rs, false, false);
    }

    // ---- fused epilogue: (acc - rowsum*centers)/N, D layout M=v+8h, N=row.
    // acc_rs[v] already holds rowsum(k0+v+8h) in this very lane.
    const float invN = 1.0f / (float)VN;
#pragma unroll
    for (int v = 0; v < 8; ++v) {
        const int k = k0 + v + 8 * h;
        const int c = c0 + row;
        out[((size_t)b * VK + k) * VC + c] =
            (acc[v] - acc_rs[v] * centers[(size_t)k * VC + c]) * invN;
    }
}

extern "C" void kernel_launch(void* const* d_in, const int* in_sizes, int n_in,
                              void* d_out, int out_size, void* d_ws, size_t ws_size,
                              hipStream_t stream) {
    const float* x       = (const float*)d_in[0];  // (16,32,14,14,512)
    const float* kernel  = (const float*)d_in[1];  // (1,1,1,512,64) == (C,K)
    const float* bias    = (const float*)d_in[2];  // (64)
    const float* centers = (const float*)d_in[3];  // (64,512)
    float* out = (float*)d_out;                    // (16, 64*512) fp32

    __bf16* assgn_t = (__bf16*)d_ws;               // needs B*K*N*2 = 12.25 MB

    // Pass 1: 16*392 = 6272 waves -> 784 blocks of 256 threads (8 waves)
    vlad_pass1<<<dim3(VB * NT16 / 8), dim3(256), 0, stream>>>(x, kernel, bias, assgn_t);

    // Pass 2: 16*4*32 = 2048 waves -> 256 blocks of 256 threads
    vlad_pass2<<<dim3(2048 / 8), dim3(256), 0, stream>>>(x, assgn_t, centers, out);
}
